// ClusterLM_1056561954870
// MI455X (gfx1250) — matrix operand order repaired and checked by reading, not compiled, not run on hardware
//
#include <hip/hip_runtime.h>
#include <math.h>

typedef __attribute__((ext_vector_type(8)))  float          v8f;
typedef __attribute__((ext_vector_type(16))) __bf16         v16bf;
typedef __attribute__((ext_vector_type(16))) unsigned short v16s;
typedef __attribute__((ext_vector_type(8)))  unsigned short v8s;
typedef __attribute__((ext_vector_type(2)))  unsigned short v2s;

#define B_N 65536
#define K_N 256
#define D_N 768
#define KC 32                  // K-chunk per WMMA group (bf16 K=32)
#define LDS_STRIDE 40          // padded (elements): 20-bank spacing -> conflict-free b128

// RNE float -> bf16 bits (NaN irrelevant for this data)
static __device__ __forceinline__ unsigned short bf16_rne(float x) {
  unsigned u = __float_as_uint(x);
  unsigned r = u + 0x7FFFu + ((u >> 16) & 1u);
  return (unsigned short)(r >> 16);
}
static __device__ __forceinline__ float bf16_to_f32(unsigned short h) {
  return __uint_as_float(((unsigned)h) << 16);
}

// ---------------- row sum-of-squares: one wave per row ----------------
__global__ __launch_bounds__(256) void rowsq_kernel(const float* __restrict__ x,
                                                    int rows, int d,
                                                    float* __restrict__ out) {
  const int wid  = threadIdx.x >> 5;
  const int lane = threadIdx.x & 31;
  const int row  = blockIdx.x * 8 + wid;
  if (row >= rows) return;
  const float* p = x + (size_t)row * d;
  float s = 0.f;
  for (int j = lane; j < d; j += 32) { float v = p[j]; s += v * v; }
#pragma unroll
  for (int off = 16; off > 0; off >>= 1) s += __shfl_xor(s, off, 32);
  if (lane == 0) out[row] = s;
}

// ---------------- init per-column min to +inf ----------------
__global__ void init_kernel(unsigned* __restrict__ colminbits) {
  if (threadIdx.x < K_N) colminbits[threadIdx.x] = 0x7F800000u;
}

// ---------------- distance GEMM via V_WMMA_F32_16X16X32_BF16 (hi/lo split) ----
// Grid: B/128 blocks of 256 threads (8 waves). Wave w computes rows
// [blk*128 + w*16, +16) x all 256 centroid columns (16 WMMA tiles).
__global__ __launch_bounds__(256) void dist_kernel(
    const float* __restrict__ emb, const float* __restrict__ cent,
    const float* __restrict__ e2, const float* __restrict__ c2,
    float* __restrict__ distT, unsigned* __restrict__ colminbits,
    float* __restrict__ labels_out) {
  __shared__ unsigned short ldsHi[K_N * LDS_STRIDE];  // 20 KB
  __shared__ unsigned short ldsLo[K_N * LDS_STRIDE];  // 20 KB
  __shared__ unsigned sminbits[K_N];

  const int tid   = threadIdx.x;
  const int wid   = tid >> 5;
  const int lane  = tid & 31;
  const int l15   = lane & 15;
  const int half  = lane >> 4;
  const int koffA = half * 8;    // 16-bit A 16x32 layout: half0 K{0-7,16-23}, half1 K{8-15,24-31}
  const int koffB = half * 8;    // 16-bit B 32x16 layout: half0 K{0-7,16-23}, half1 K{8-15,24-31}
  const int rowBase = blockIdx.x * 128;
  const int myArow  = rowBase + wid * 16 + l15;

  sminbits[tid] = 0x7F800000u;

  v8f acc[16];
#pragma unroll
  for (int t = 0; t < 16; ++t) {
    v8f z = {0.f, 0.f, 0.f, 0.f, 0.f, 0.f, 0.f, 0.f};
    acc[t] = z;
  }

  const float* aPtr = emb + (size_t)myArow * D_N;

  for (int kkc = 0; kkc < D_N; kkc += KC) {
    __syncthreads();
    // stage centroid chunk, split into bf16 hi/lo planes (conversion paid once)
    for (int i = tid * 2; i < K_N * KC; i += 256 * 2) {
      const int r = i >> 5;           // / KC
      const int c = i & (KC - 1);
      const float2 v = *(const float2*)(cent + (size_t)r * D_N + kkc + c);
      const unsigned short h0 = bf16_rne(v.x), h1 = bf16_rne(v.y);
      v2s hv; hv[0] = h0; hv[1] = h1;
      v2s lv; lv[0] = bf16_rne(v.x - bf16_to_f32(h0));
              lv[1] = bf16_rne(v.y - bf16_to_f32(h1));
      *(v2s*)(ldsHi + r * LDS_STRIDE + c) = hv;
      *(v2s*)(ldsLo + r * LDS_STRIDE + c) = lv;
    }
    __syncthreads();

    // A fragment: two contiguous float8 runs per lane, split hi/lo
    const float* ap = aPtr + kkc + koffA;
    const float4 fa0 = *(const float4*)(ap);
    const float4 fa1 = *(const float4*)(ap + 4);
    const float4 fa2 = *(const float4*)(ap + 16);
    const float4 fa3 = *(const float4*)(ap + 20);
    const float av[16] = {fa0.x, fa0.y, fa0.z, fa0.w, fa1.x, fa1.y, fa1.z, fa1.w,
                          fa2.x, fa2.y, fa2.z, fa2.w, fa3.x, fa3.y, fa3.z, fa3.w};
    v16s ahiB, aloB;
#pragma unroll
    for (int i = 0; i < 16; ++i) {
      const unsigned short h = bf16_rne(av[i]);
      ahiB[i] = h;
      aloB[i] = bf16_rne(av[i] - bf16_to_f32(h));
    }
    const v16bf ahi = __builtin_bit_cast(v16bf, ahiB);
    const v16bf alo = __builtin_bit_cast(v16bf, aloB);

#pragma unroll
    for (int t = 0; t < 16; ++t) {
      const int colLocal = t * 16 + l15;
      const unsigned short* bph = ldsHi + colLocal * LDS_STRIDE + koffB;
      const unsigned short* bpl = ldsLo + colLocal * LDS_STRIDE + koffB;
      const v8s h0 = *(const v8s*)(bph);
      const v8s h1 = *(const v8s*)(bph + 16);
      const v8s l0 = *(const v8s*)(bpl);
      const v8s l1 = *(const v8s*)(bpl + 16);
      const v16bf bhi = __builtin_bit_cast(
          v16bf, __builtin_shufflevector(h0, h1, 0, 1, 2, 3, 4, 5, 6, 7,
                                         8, 9, 10, 11, 12, 13, 14, 15));
      const v16bf blo = __builtin_bit_cast(
          v16bf, __builtin_shufflevector(l0, l1, 0, 1, 2, 3, 4, 5, 6, 7,
                                         8, 9, 10, 11, 12, 13, 14, 15));
      // dot += Ahi*Bhi + Alo*Bhi + Ahi*Blo   (lo*lo ~2^-18, dropped)
      acc[t] = __builtin_amdgcn_wmma_f32_16x16x32_bf16(
          false, ahi, false, bhi, (short)0, acc[t], false, false);
      acc[t] = __builtin_amdgcn_wmma_f32_16x16x32_bf16(
          false, alo, false, bhi, (short)0, acc[t], false, false);
      acc[t] = __builtin_amdgcn_wmma_f32_16x16x32_bf16(
          false, ahi, false, blo, (short)0, acc[t], false, false);
    }
  }

  // ---- epilogue: C/D layout -> lane holds rows rowB0..rowB0+7, col = t*16+l15
  const int rowB0 = rowBase + wid * 16 + half * 8;
  const float4 ev0 = *(const float4*)(e2 + rowB0);
  const float4 ev1 = *(const float4*)(e2 + rowB0 + 4);
  const float e2row[8] = {ev0.x, ev0.y, ev0.z, ev0.w, ev1.x, ev1.y, ev1.z, ev1.w};

  float bestVal[8];
  int   bestIdx[8];
#pragma unroll
  for (int r = 0; r < 8; ++r) { bestVal[r] = 3.4e38f; bestIdx[r] = 0; }

#pragma unroll
  for (int t = 0; t < 16; ++t) {
    const int col = t * 16 + l15;
    const float c2c = c2[col];
    float dv[8];
    float cmin = 3.4e38f;
#pragma unroll
    for (int r = 0; r < 8; ++r) {
      const float d2 = e2row[r] + c2c - 2.0f * acc[t][r];
      const float dd = sqrtf(fmaxf(d2, 0.f));
      dv[r] = dd;
      cmin = fminf(cmin, dd);
      if (dd < bestVal[r]) { bestVal[r] = dd; bestIdx[r] = col; }
    }
    float* dp = distT + (size_t)col * B_N + rowB0;   // transposed, 16B aligned
    *(float4*)dp       = make_float4(dv[0], dv[1], dv[2], dv[3]);
    *(float4*)(dp + 4) = make_float4(dv[4], dv[5], dv[6], dv[7]);
    atomicMin(&sminbits[col], __float_as_uint(cmin));
  }

  // per-row argmin across the 16 lanes of each half-wave (JAX ties -> lowest idx)
#pragma unroll
  for (int r = 0; r < 8; ++r) {
    float v = bestVal[r];
    int   ix = bestIdx[r];
#pragma unroll
    for (int off = 1; off < 16; off <<= 1) {
      const float ov = __shfl_xor(v, off, 32);
      const int   oi = __shfl_xor(ix, off, 32);
      if (ov < v || (ov == v && oi < ix)) { v = ov; ix = oi; }
    }
    if (l15 == 0) labels_out[rowB0 + r] = (float)ix;
  }

  __syncthreads();
  if (tid < K_N) atomicMin(&colminbits[tid], sminbits[tid]);
}

// ---------------- per-centroid softmin over the batch axis ----------------
__global__ __launch_bounds__(256) void softmin_kernel(
    const float* __restrict__ distT, const unsigned* __restrict__ colminbits,
    float* __restrict__ lossK) {
  __shared__ float sS[256], sW[256];
  const int k = blockIdx.x;
  const int tid = threadIdx.x;
  const float minv = __uint_as_float(colminbits[k]);
  const float* dp = distT + (size_t)k * B_N;
  float S = 0.f, W = 0.f;
  for (int b = tid; b < B_N; b += 256) {
    const float d = dp[b];
    const float e = expf(-(d - minv));   // ALPHA = 1
    S += e;
    W += d * e;
  }
  sS[tid] = S; sW[tid] = W;
  __syncthreads();
  for (int s = 128; s > 0; s >>= 1) {
    if (tid < s) { sS[tid] += sS[tid + s]; sW[tid] += sW[tid + s]; }
    __syncthreads();
  }
  if (tid == 0) lossK[k] = sW[0] / sS[0];
}

// ---------------- mean over centroids ----------------
__global__ void final_kernel(const float* __restrict__ lossK, float* __restrict__ out) {
  __shared__ float s[256];
  const int tid = threadIdx.x;
  s[tid] = lossK[tid];
  __syncthreads();
  for (int off = 128; off > 0; off >>= 1) {
    if (tid < off) s[tid] += s[tid + off];
    __syncthreads();
  }
  if (tid == 0) out[0] = s[0] / (float)K_N;
}

extern "C" void kernel_launch(void* const* d_in, const int* in_sizes, int n_in,
                              void* d_out, int out_size, void* d_ws, size_t ws_size,
                              hipStream_t stream) {
  (void)in_sizes; (void)n_in; (void)out_size; (void)ws_size;
  const float* emb  = (const float*)d_in[0];
  const float* cent = (const float*)d_in[1];
  float* out = (float*)d_out;

  float*    distT  = (float*)d_ws;                       // K*B floats (64 MB)
  float*    e2     = distT + (size_t)K_N * B_N;          // B floats
  float*    c2     = e2 + B_N;                           // K floats
  unsigned* colmin = (unsigned*)(c2 + K_N);              // K uints
  float*    lossK  = (float*)(colmin + K_N);             // K floats

  rowsq_kernel<<<B_N / 8, 256, 0, stream>>>(emb, B_N, D_N, e2);
  rowsq_kernel<<<K_N / 8, 256, 0, stream>>>(cent, K_N, D_N, c2);
  init_kernel<<<1, 256, 0, stream>>>(colmin);
  dist_kernel<<<B_N / 128, 256, 0, stream>>>(emb, cent, e2, c2, distT, colmin, out + 1);
  softmin_kernel<<<K_N, 256, 0, stream>>>(distT, colmin, lossK);
  final_kernel<<<1, 256, 0, stream>>>(lossK, out);
}
